// QuantLinear_6485400616973
// MI455X (gfx1250) — compile-verified, matrix operand order checked by reading, not run on hardware
//
#include <hip/hip_runtime.h>
#include <cstdint>

typedef __attribute__((ext_vector_type(16))) _Float16 v16h;
typedef __attribute__((ext_vector_type(8)))  _Float16 v8h;
typedef __attribute__((ext_vector_type(8)))  float    v8f;
typedef __attribute__((ext_vector_type(4)))  float    v4f;
typedef __attribute__((ext_vector_type(4)))  int      v4i;

#define K_DIM     4096
#define N_DIM     11008
#define M_DIM     4096      // B*S = 2*2048
#define GROUPSZ   128

#define BM        128
#define BN        128
#define BK        64
#define NSTAGE    (K_DIM / BK)
#define LDA       (BK + 8)  // padded halfs per LDS row (144B, 16B-aligned)
#define LDB       (BK + 8)
#define A_HALFS   (BM * LDA)
#define B_HALFS   (BN * LDB)

// ---- gfx1250 async memory->LDS path (guarded; sync fallback otherwise) ----
#if defined(__has_builtin)
#if __has_builtin(__builtin_amdgcn_global_load_async_to_lds_b128) && \
    __has_builtin(__builtin_amdgcn_s_wait_asynccnt)
#define HAVE_ASYNC_LDS 1
#endif
#endif
#ifndef HAVE_ASYNC_LDS
#define HAVE_ASYNC_LDS 0
#endif

#if HAVE_ASYNC_LDS
typedef __attribute__((address_space(1))) v4i gv4i_t;   // global int4
typedef __attribute__((address_space(3))) v4i lv4i_t;   // LDS int4
__device__ __forceinline__ void stage16_async(const _Float16* g, _Float16* l) {
    // generic->as(1)/as(3) via integer round-trip (low 32 bits == LDS address)
    __builtin_amdgcn_global_load_async_to_lds_b128(
        (gv4i_t*)(uintptr_t)g, (lv4i_t*)(uintptr_t)l, 0, 0);
}
#define WAIT_ASYNC() __builtin_amdgcn_s_wait_asynccnt(0)
#else
__device__ __forceinline__ void stage16_async(const _Float16* g, _Float16* l) {
    *(v8h*)l = *(const v8h*)g;
}
#define WAIT_ASYNC() ((void)0)
#endif

// ---------------------------------------------------------------------------
// Pre-pass: x fp32 -> fp16 (once; result is L2-resident, halves A traffic)
// ---------------------------------------------------------------------------
__global__ __launch_bounds__(256) void xcvt_kernel(const float* __restrict__ x,
                                                   _Float16* __restrict__ xh) {
    int i = (blockIdx.x * 256 + threadIdx.x) * 8;
    v4f a = *(const v4f*)(x + i);
    v4f b = *(const v4f*)(x + i + 4);
    v8h h;
    h[0] = (_Float16)a[0]; h[1] = (_Float16)a[1];
    h[2] = (_Float16)a[2]; h[3] = (_Float16)a[3];
    h[4] = (_Float16)b[0]; h[5] = (_Float16)b[1];
    h[6] = (_Float16)b[2]; h[7] = (_Float16)b[3];
    *(v8h*)(xh + i) = h;
}

// ---------------------------------------------------------------------------
// GPTQ int4 GEMM: out[M,N] = xh[M,K] * dequant(qweight)[K,N] + bias
// 256 threads = 8 waves; 128x128 tile; wave = 32(M) x 64(N) = 2x4 WMMA tiles.
// Double-buffered LDS; A staged by async-to-LDS DMA; B dequant pipelined
// through registers. One barrier per K-stage.
// ---------------------------------------------------------------------------
__global__ __launch_bounds__(256) void gptq_wmma_kernel(
    const _Float16* __restrict__ xh,
    const int*      __restrict__ qweight,   // [K/8, N] (8 K-nibbles per int32)
    const int*      __restrict__ qzeros,    // [NG, N/8] (8 N-nibbles per int32)
    const float*    __restrict__ scales,    // [NG, N]
    const float*    __restrict__ bias,      // [N]
    float*          __restrict__ out)       // [M, N]
{
    __shared__ _Float16 smem[2][A_HALFS + B_HALFS];   // 2 x 36 KB

    const int tid  = threadIdx.x;
    const int lane = tid & 31;
    const int wave = tid >> 5;
    const int wm   = wave & 3;              // 4 waves along M (32 rows each)
    const int wn   = wave >> 2;             // 2 waves along N (64 cols each)
    const int m0   = blockIdx.y * BM;
    const int n0   = blockIdx.x * BN;

    const int lm = lane & 15;               // row/col within 16x16 fragment
    const int lk = (lane >> 4) * 8;         // K-half-select per WMMA A/B layout

    v8f acc[2][4];
#pragma unroll
    for (int mi = 0; mi < 2; ++mi)
#pragma unroll
        for (int nj = 0; nj < 4; ++nj)
            acc[mi][nj] = (v8f){0.f, 0.f, 0.f, 0.f, 0.f, 0.f, 0.f, 0.f};

    // --- per-thread staging descriptors (4 x 16B chunks per tile each) ---
    // A: chunk c -> row r = c>>3, k-octet ko = c&7
    // B: chunk c -> col n = c&127, k-octet ko = c>>7 (consecutive lanes -> consecutive n)
    int   qreg[4];
    int   zreg[4];
    float sreg[4];

    // Issue A-tile async copy for stage `k0` into buffer `As`
    auto issue_a = [&](int k0, _Float16* As) {
#pragma unroll
        for (int i = 0; i < 4; ++i) {
            int c = tid + i * 256;
            int r = c >> 3, ko = c & 7;
            stage16_async(xh + (size_t)(m0 + r) * K_DIM + k0 + ko * 8,
                          &As[r * LDA + ko * 8]);
        }
    };

    // Load qweight/zero/scale for stage `k0` into registers (no LDS yet)
    auto fetch_b = [&](int k0) {
        const int g = k0 >> 7;              // BK=64 < GROUPSZ=128: one group/stage
#pragma unroll
        for (int i = 0; i < 4; ++i) {
            int c  = tid + i * 256;
            int n  = c & (BN - 1), ko = c >> 7;
            int gn = n0 + n;
            qreg[i] = qweight[(size_t)((k0 >> 3) + ko) * N_DIM + gn];
            int zq  = qzeros[(size_t)g * (N_DIM >> 3) + (gn >> 3)];
            zreg[i] = ((zq >> ((gn & 7) * 4)) & 0xF) + 1;   // GPTQ stores zero-1
            sreg[i] = scales[(size_t)g * N_DIM + gn];
        }
    };

    // Dequant registers -> LDS B^T tile
    auto store_b = [&](_Float16* Bs) {
#pragma unroll
        for (int i = 0; i < 4; ++i) {
            int c = tid + i * 256;
            int n = c & (BN - 1), ko = c >> 7;
            v8h w;
#pragma unroll
            for (int j = 0; j < 8; ++j) {
                int qj = (qreg[i] >> (j * 4)) & 0xF;
                w[j] = (_Float16)((float)(qj - zreg[i]) * sreg[i]);
            }
            *(v8h*)(&Bs[n * LDB + ko * 8]) = w;
        }
    };

    // ---- pipeline prologue: stage 0 in flight ----
    issue_a(0, &smem[0][0]);
    fetch_b(0);

    for (int s = 0; s < NSTAGE; ++s) {
        const int cur = s & 1, nxt = cur ^ 1;
        _Float16* As = &smem[cur][0];
        _Float16* Bs = &smem[cur][A_HALFS];

        // finish stage s: dequant B from regs, wait for async A arrival
        store_b(Bs);
        WAIT_ASYNC();
        __syncthreads();        // buf[cur] ready for all; buf[nxt] free for all

        // kick off stage s+1 while stage s computes
        if (s + 1 < NSTAGE) {
            issue_a((s + 1) * BK, &smem[nxt][0]);
            fetch_b((s + 1) * BK);
        }

        // ---- two k=32 WMMA steps over the 64-deep stage ----
#pragma unroll
        for (int kk = 0; kk < BK; kk += 32) {
            v16h a[2];
#pragma unroll
            for (int mi = 0; mi < 2; ++mi) {
                const _Float16* p = &As[(wm * 32 + mi * 16 + lm) * LDA + kk + lk];
                v8h lo = *(const v8h*)p;          // K = kk+lk   .. +7
                v8h hi = *(const v8h*)(p + 16);   // K = kk+lk+16 .. +7
                a[mi] = __builtin_shufflevector(lo, hi, 0, 1, 2, 3, 4, 5, 6, 7,
                                                8, 9, 10, 11, 12, 13, 14, 15);
            }
            v16h b[4];
#pragma unroll
            for (int nj = 0; nj < 4; ++nj) {
                const _Float16* p = &Bs[(wn * 64 + nj * 16 + lm) * LDB + kk + lk];
                v8h lo = *(const v8h*)p;
                v8h hi = *(const v8h*)(p + 16);
                b[nj] = __builtin_shufflevector(lo, hi, 0, 1, 2, 3, 4, 5, 6, 7,
                                                8, 9, 10, 11, 12, 13, 14, 15);
            }
#pragma unroll
            for (int mi = 0; mi < 2; ++mi)
#pragma unroll
                for (int nj = 0; nj < 4; ++nj)
                    acc[mi][nj] = __builtin_amdgcn_wmma_f32_16x16x32_f16(
                        false, a[mi], false, b[nj],
                        (short)0, acc[mi][nj], false, false);
        }
        // no trailing barrier: next iteration's barrier protects buffer reuse
    }

    // ---- Epilogue: bias add + store per documented C/D layout ----
#pragma unroll
    for (int mi = 0; mi < 2; ++mi) {
        int rbase = m0 + wm * 32 + mi * 16 + ((lane >> 4) << 3);
#pragma unroll
        for (int nj = 0; nj < 4; ++nj) {
            int col  = n0 + wn * 64 + nj * 16 + lm;
            float bv = bias[col];
            v8f c = acc[mi][nj];
#pragma unroll
            for (int i = 0; i < 8; ++i)
                out[(size_t)(rbase + i) * N_DIM + col] = c[i] + bv;
        }
    }
}

// ---------------------------------------------------------------------------
extern "C" void kernel_launch(void* const* d_in, const int* in_sizes, int n_in,
                              void* d_out, int out_size, void* d_ws, size_t ws_size,
                              hipStream_t stream) {
    const float* x       = (const float*)d_in[0];   // [2,2048,4096] fp32
    const int*   qweight = (const int*)d_in[1];     // [512, 11008]
    const int*   qzeros  = (const int*)d_in[2];     // [32, 1376]
    const float* scales  = (const float*)d_in[3];   // [32, 11008]
    // d_in[4] = g_idx: implied by k/GROUPSZ, unused
    const float* bias    = (const float*)d_in[5];   // [11008]
    float*       out     = (float*)d_out;           // [4096, 11008]

    _Float16* xh = (_Float16*)d_ws;                 // 33.5 MB fp16 activations

    // x fp32 -> fp16 (16,777,216 elements, 8 per thread)
    xcvt_kernel<<<(M_DIM * K_DIM) / (256 * 8), 256, 0, stream>>>(x, xh);

    dim3 grid(N_DIM / BN, M_DIM / BM);              // 86 x 32
    gptq_wmma_kernel<<<grid, 256, 0, stream>>>(xh, qweight, qzeros, scales, bias, out);
}
